// GAttnBlock_13099650253561
// MI455X (gfx1250) — matrix operand order repaired and checked by reading, not compiled, not run on hardware
//
#include <hip/hip_runtime.h>
#include <hip/hip_bf16.h>
#include <stdint.h>

#define NTOK 8192
#define DF   512

typedef __bf16 v16bf __attribute__((ext_vector_type(16)));
typedef float  v8f   __attribute__((ext_vector_type(8)));

union V16 { v16bf v; uint4 q[2]; unsigned short h[16]; };

__device__ __forceinline__ unsigned short f2bf(float f) {
    unsigned u = __float_as_uint(f);
    u += 0x7FFFu + ((u >> 16) & 1u);
    return (unsigned short)(u >> 16);
}

__device__ __forceinline__ v8f wmma_bf16(v16bf a, v16bf b, v8f c) {
    return __builtin_amdgcn_wmma_f32_16x16x32_bf16(false, a, false, b, (short)0, c, false, false);
}

// Async global->LDS copy (gfx1250 VGLOBAL async encoding, tracked by ASYNCcnt).
// lds_off: byte offset within LDS (low 32 bits of a flat shared pointer).
__device__ __forceinline__ void async_g2l_b128(unsigned lds_off, const void* gaddr) {
    asm volatile("global_load_async_to_lds_b128 %0, %1, off"
                 :: "v"(lds_off), "v"(gaddr) : "memory");
}
__device__ __forceinline__ void wait_async0() {
    asm volatile("s_wait_asynccnt 0x0" ::: "memory");
}

// ---------------- f32 -> bf16 weight conversion ----------------
__global__ void wconv(const float* __restrict__ w, unsigned short* __restrict__ o, int n) {
    int i = blockIdx.x * blockDim.x + threadIdx.x;
    if (i < n) o[i] = f2bf(w[i]);
}

// ---------------- LayerNorm (one row per block) ----------------
__global__ void lnorm(const float* __restrict__ x, const float* __restrict__ g,
                      const float* __restrict__ b, unsigned short* __restrict__ h) {
    __shared__ float s1[256];
    __shared__ float s2[256];
    int row = blockIdx.x, tid = threadIdx.x;
    float2 xv = *(const float2*)&x[row * DF + tid * 2];
    s1[tid] = xv.x + xv.y;
    s2[tid] = xv.x * xv.x + xv.y * xv.y;
    __syncthreads();
    for (int off = 128; off > 0; off >>= 1) {
        if (tid < off) { s1[tid] += s1[tid + off]; s2[tid] += s2[tid + off]; }
        __syncthreads();
    }
    float mu  = s1[0] * (1.0f / DF);
    float var = s2[0] * (1.0f / DF) - mu * mu;
    float rs  = rsqrtf(var + 1e-5f);
    float2 gv = *(const float2*)&g[tid * 2];
    float2 bv = *(const float2*)&b[tid * 2];
    float h0 = (xv.x - mu) * rs * gv.x + bv.x;
    float h1 = (xv.y - mu) * rs * gv.y + bv.y;
    unsigned pack = (unsigned)f2bf(h0) | ((unsigned)f2bf(h1) << 16);
    ((unsigned*)h)[row * (DF / 2) + tid] = pack;
}

// ---------------- WMMA GEMM:  C = A[8192,512](bf16) @ W[512,512]^T(bf16) + bias ----------------
// mode 0: out = bf16 row-major, val = (acc+bias)*scale       (Q with 1/sqrt(d), K)
// mode 1: out = bf16 TRANSPOSED store [512, 8192]            (V^T)
// mode 2: out = f32 row-major, val = resid + acc + bias      (final projection + residual)
__global__ void __launch_bounds__(256) gemm(const unsigned short* __restrict__ A,
                                            const unsigned short* __restrict__ W,
                                            const float* __restrict__ bias,
                                            unsigned short* __restrict__ outb,
                                            float* __restrict__ outf,
                                            const float* __restrict__ resid,
                                            float scale, int mode) {
    int tid = threadIdx.x;
    int w = tid >> 5, lane = tid & 31, lo = lane & 15, hi = lane >> 4;
    int r0 = blockIdx.x * 16;
    int colbase = w * 64;
    v8f acc[4] = {};
    const unsigned short* arow = A + (r0 + lo) * DF;
    #pragma unroll
    for (int kc = 0; kc < 16; ++kc) {
        int kb = kc * 32;
        V16 a;
        a.q[0] = *(const uint4*)(arow + kb + hi * 16);
        a.q[1] = *(const uint4*)(arow + kb + 8 + hi * 16);
        v16bf bm[4];
        #pragma unroll
        for (int t = 0; t < 4; ++t)
            bm[t] = *(const v16bf*)(W + (colbase + t * 16 + lo) * DF + kb + hi * 16);
        #pragma unroll
        for (int t = 0; t < 4; ++t)
            acc[t] = wmma_bf16(a.v, bm[t], acc[t]);
    }
    #pragma unroll
    for (int t = 0; t < 4; ++t) {
        int col = colbase + t * 16 + lo;
        float bv = bias[col];
        #pragma unroll
        for (int rr = 0; rr < 8; ++rr) {
            int row = r0 + rr + 8 * hi;
            float val = acc[t][rr] + bv;
            if (mode == 2) {
                outf[row * DF + col] = resid[row * DF + col] + val;
            } else if (mode == 1) {
                outb[col * NTOK + row] = f2bf(val);
            } else {
                outb[row * DF + col] = f2bf(val * scale);
            }
        }
    }
}

// ---------------- single-pass flash attention ----------------
// Block: 256 threads = 8 waves = 2 query-row-groups (16 rows each) x 4 key-slice waves.
// Per iteration: 64 keys. wave(qy,wx): S tile 16x16 via 16 WMMAs, online softmax
// (cross-wave max/sum via LDS), P(bf16) staged in LDS, then P @ V^T via 16 WMMAs.
__global__ void __launch_bounds__(256) attn(const unsigned short* __restrict__ Q,
                                            const unsigned short* __restrict__ K,
                                            const unsigned short* __restrict__ VT,
                                            unsigned short* __restrict__ O) {
    __shared__ __align__(16) unsigned short qs[32 * DF];       // 32 KB query tile
    __shared__ __align__(16) unsigned short ptile[2][16 * 64]; // 4 KB P tiles
    __shared__ __align__(16) float mbuf[2][16][4];
    __shared__ __align__(16) float sbuf[2][16][4];

    int tid = threadIdx.x;
    int w = tid >> 5, lane = tid & 31, lo = lane & 15, hi = lane >> 4;
    int qy = w >> 2, wx = w & 3;
    int q0 = blockIdx.x * 32;

    // stage the 32x512 bf16 query tile into LDS via async global->LDS DMA
    {
        const unsigned short* src = Q + q0 * DF;
        #pragma unroll
        for (int rep = 0; rep < 8; ++rep) {
            int i = tid + rep * 256;              // 2048 x b128 total
            unsigned lds_off = (unsigned)(unsigned long long)(const void*)(qs + i * 8);
            async_g2l_b128(lds_off, src + i * 8);
        }
        wait_async0();
    }
    __syncthreads();

    float m[8], l[8];
    v8f acc[8] = {};
    #pragma unroll
    for (int rr = 0; rr < 8; ++rr) { m[rr] = -3.0e38f; l[rr] = 0.0f; }

    const unsigned short* qrow = qs + (qy * 16 + lo) * DF;

    for (int kb = 0; kb < NTOK / 64; ++kb) {
        int key0 = kb * 64 + wx * 16;
        // ---- S = Q * K^T  (16 chained WMMAs over d=512) ----
        v8f s = {};
        #pragma unroll
        for (int kc = 0; kc < 16; ++kc) {
            int kk = kc * 32;
            V16 a;
            a.q[0] = *(const uint4*)(qrow + kk + hi * 16);
            a.q[1] = *(const uint4*)(qrow + kk + 8 + hi * 16);
            v16bf bm = *(const v16bf*)(K + (key0 + lo) * DF + kk + hi * 16);
            s = wmma_bf16(a.v, bm, s);
        }
        // prefetch next iteration's K rows and V^T lines (global_prefetch_b8)
        if (kb + 1 < NTOK / 64) {
            __builtin_prefetch(K + (key0 + 64 + lo) * DF, 0, 3);
            __builtin_prefetch(VT + (wx * 128 + lo) * NTOK + (kb + 1) * 64, 0, 3);
        }
        // ---- per-wave row max over 16-key slice ----
        float tm[8];
        #pragma unroll
        for (int rr = 0; rr < 8; ++rr) {
            float t = s[rr];
            t = fmaxf(t, __shfl_xor(t, 1));
            t = fmaxf(t, __shfl_xor(t, 2));
            t = fmaxf(t, __shfl_xor(t, 4));
            t = fmaxf(t, __shfl_xor(t, 8));
            tm[rr] = t;
        }
        if (lo == 0) {
            #pragma unroll
            for (int rr = 0; rr < 8; ++rr) mbuf[qy][rr + 8 * hi][wx] = tm[rr];
        }
        __syncthreads();
        // ---- online softmax update (block max over all 64 keys) ----
        #pragma unroll
        for (int rr = 0; rr < 8; ++rr) {
            float4 mv = *(const float4*)&mbuf[qy][rr + 8 * hi][0];
            float bm4 = fmaxf(fmaxf(mv.x, mv.y), fmaxf(mv.z, mv.w));
            float mn = fmaxf(m[rr], bm4);
            float es = __expf(m[rr] - mn);
            m[rr] = mn;
            l[rr] = l[rr] * es;
            #pragma unroll
            for (int t = 0; t < 8; ++t) acc[t][rr] = acc[t][rr] * es;
            float p = __expf(s[rr] - mn);
            float rs = p;
            rs += __shfl_xor(rs, 1);
            rs += __shfl_xor(rs, 2);
            rs += __shfl_xor(rs, 4);
            rs += __shfl_xor(rs, 8);
            if (lo == 0) sbuf[qy][rr + 8 * hi][wx] = rs;
            ptile[qy][(rr + 8 * hi) * 64 + wx * 16 + lo] = f2bf(p);
        }
        __syncthreads();
        #pragma unroll
        for (int rr = 0; rr < 8; ++rr) {
            float4 sv = *(const float4*)&sbuf[qy][rr + 8 * hi][0];
            l[rr] += sv.x + sv.y + sv.z + sv.w;
        }
        // ---- O += P @ V  (wave covers 128 output features, 2 k-steps of 32 keys) ----
        #pragma unroll
        for (int k2 = 0; k2 < 2; ++k2) {
            const unsigned short* prow = &ptile[qy][lo * 64 + k2 * 32];
            V16 a;
            a.q[0] = *(const uint4*)(prow + hi * 16);
            a.q[1] = *(const uint4*)(prow + 8 + hi * 16);
            #pragma unroll
            for (int th = 0; th < 2; ++th) {
                v16bf bm[4];
                #pragma unroll
                for (int t = 0; t < 4; ++t) {
                    int feat = wx * 128 + (th * 4 + t) * 16 + lo;
                    bm[t] = *(const v16bf*)(VT + feat * NTOK + kb * 64 + k2 * 32 + hi * 16);
                }
                #pragma unroll
                for (int t = 0; t < 4; ++t)
                    acc[th * 4 + t] = wmma_bf16(a.v, bm[t], acc[th * 4 + t]);
            }
        }
        __syncthreads();
    }
    // ---- normalize by l and store bf16 ----
    #pragma unroll
    for (int rr = 0; rr < 8; ++rr) {
        float inv = 1.0f / l[rr];
        int row = q0 + qy * 16 + rr + 8 * hi;
        #pragma unroll
        for (int t = 0; t < 8; ++t) {
            int feat = wx * 128 + t * 16 + lo;
            O[row * DF + feat] = f2bf(acc[t][rr] * inv);
        }
    }
}

extern "C" void kernel_launch(void* const* d_in, const int* in_sizes, int n_in,
                              void* d_out, int out_size, void* d_ws, size_t ws_size,
                              hipStream_t stream) {
    (void)in_sizes; (void)n_in; (void)out_size; (void)ws_size;
    const float* x    = (const float*)d_in[0];
    const float* ln_w = (const float*)d_in[1];
    const float* ln_b = (const float*)d_in[2];
    const float* Wq   = (const float*)d_in[3];
    const float* bq   = (const float*)d_in[4];
    const float* Wk   = (const float*)d_in[5];
    const float* bk   = (const float*)d_in[6];
    const float* Wv   = (const float*)d_in[7];
    const float* bv   = (const float*)d_in[8];
    const float* Wp   = (const float*)d_in[9];
    const float* bp   = (const float*)d_in[10];
    float* out = (float*)d_out;

    char* ws = (char*)d_ws;
    unsigned short* h_bf  = (unsigned short*)(ws);                          // 8 MB
    unsigned short* q_bf  = (unsigned short*)(ws + (size_t)(8 << 20));      // 8 MB
    unsigned short* k_bf  = (unsigned short*)(ws + (size_t)(16 << 20));     // 8 MB
    unsigned short* vT_bf = (unsigned short*)(ws + (size_t)(24 << 20));     // 8 MB  [DF, NTOK]
    unsigned short* a_bf  = (unsigned short*)(ws + (size_t)(32 << 20));     // 8 MB
    unsigned short* wq_bf = (unsigned short*)(ws + (size_t)(40 << 20));
    unsigned short* wk_bf = (unsigned short*)(ws + (size_t)(40 << 20) + (1 << 19));
    unsigned short* wv_bf = (unsigned short*)(ws + (size_t)(40 << 20) + (2 << 19));
    unsigned short* wp_bf = (unsigned short*)(ws + (size_t)(40 << 20) + (3 << 19));

    const int wn = DF * DF;
    wconv<<<wn / 256, 256, 0, stream>>>(Wq, wq_bf, wn);
    wconv<<<wn / 256, 256, 0, stream>>>(Wk, wk_bf, wn);
    wconv<<<wn / 256, 256, 0, stream>>>(Wv, wv_bf, wn);
    wconv<<<wn / 256, 256, 0, stream>>>(Wp, wp_bf, wn);

    lnorm<<<NTOK, 256, 0, stream>>>(x, ln_w, ln_b, h_bf);

    const float qscale = 0.044194173824159216f; // 1/sqrt(512)
    gemm<<<NTOK / 16, 256, 0, stream>>>(h_bf, wq_bf, bq, q_bf, nullptr, nullptr, qscale, 0);
    gemm<<<NTOK / 16, 256, 0, stream>>>(h_bf, wk_bf, bk, k_bf, nullptr, nullptr, 1.0f, 0);
    gemm<<<NTOK / 16, 256, 0, stream>>>(h_bf, wv_bf, bv, vT_bf, nullptr, nullptr, 1.0f, 1);

    attn<<<NTOK / 32, 256, 0, stream>>>(q_bf, k_bf, vT_bf, a_bf);

    gemm<<<NTOK / 16, 256, 0, stream>>>(a_bf, wp_bf, bp, nullptr, out, x, 1.0f, 2);
}
